// Torso_55671366091051
// MI455X (gfx1250) — matrix operand order repaired and checked by reading, not compiled, not run on hardware
//
#include <hip/hip_runtime.h>
#include <hip/hip_bf16.h>

// ---------------------------------------------------------------------------
// CDNA5 (gfx1250): bf16 WMMA GEMMs, fp32 accumulate.
// Prologue tiles via Tensor Data Mover (TENSORcnt), steady-state K-loop tiles
// via GLOBAL_LOAD_ASYNC_TO_LDS_B128 (ASYNCcnt), compute via v_wmma bf16.
// ---------------------------------------------------------------------------

typedef __bf16 bf16_t;
typedef __attribute__((ext_vector_type(16))) __bf16 v16bf;
typedef __attribute__((ext_vector_type(8)))  float  v8f;
typedef int          v4i_g __attribute__((vector_size(16)));
typedef unsigned int v4u_g __attribute__((vector_size(16)));
typedef int          v8i_g __attribute__((vector_size(32)));

#define AS1 __attribute__((address_space(1)))
#define AS3 __attribute__((address_space(3)))

#if defined(__has_builtin)
#  if __has_builtin(__builtin_amdgcn_global_load_async_to_lds_b128)
#    define HAVE_ASYNC_LDS 1
#  endif
#  if __has_builtin(__builtin_amdgcn_tensor_load_to_lds)
#    define HAVE_TDM 1
#  endif
#endif
#ifndef HAVE_ASYNC_LDS
#  define HAVE_ASYNC_LDS 0
#endif
#ifndef HAVE_TDM
#  define HAVE_TDM 0
#endif

#define WMMA_BF16(a, b, c) \
  __builtin_amdgcn_wmma_f32_16x16x32_bf16(false, (a), false, (b), (short)0, (c), false, false)

// 16-byte tile-chunk copy global -> LDS. Async (ASYNCcnt) when available.
__device__ __forceinline__ void cp16(bf16_t* lds, const bf16_t* g) {
#if HAVE_ASYNC_LDS
  __builtin_amdgcn_global_load_async_to_lds_b128((AS1 v4i_g*)(g), (AS3 v4i_g*)(lds), 0, 0);
#else
  *(uint4*)lds = *(const uint4*)g;
#endif
}

__device__ __forceinline__ void wait_stage() {
#if HAVE_ASYNC_LDS
#  if __has_builtin(__builtin_amdgcn_s_wait_asynccnt)
  __builtin_amdgcn_s_wait_asynccnt(0);
#  else
  asm volatile("s_wait_asynccnt 0" ::: "memory");
#  endif
#endif
  __syncthreads();
}

#if HAVE_TDM
// TDM 2D tile load: rows x rowlen (bf16) from global (row stride in elems) to
// LDS, padding each 16-DWORD row with 4 DWORDs -> 40-element LDS stride.
// D# packing per CDNA5 ISA 8.3-8.5 (count=1, data_size=2B, type=2 "image").
__device__ __forceinline__ void tdm_load_2d(const bf16_t* g, unsigned lds_byte_off,
                                            int rows, int rowlen_elems,
                                            long long row_stride_elems) {
  unsigned long long ga = (unsigned long long)(size_t)g;
  v4u_g g0;
  g0[0] = 1u;                                              // count=1 (user D#)
  g0[1] = lds_byte_off;                                    // lds_addr
  g0[2] = (unsigned)(ga & 0xffffffffu);                    // global_addr[31:0]
  g0[3] = (unsigned)((ga >> 32) & 0x01ffffffu) | (2u << 30);  // [56:32] | type=2
  unsigned d0 = (1u << 16)   // data_size: 2 bytes
              | (1u << 20)   // pad_enable
              | (3u << 22)   // pad_interval: 16 DWORDs
              | (3u << 25);  // pad_amount: 4 DWORDs
  unsigned td0 = (unsigned)row_stride_elems;   // tensor_dim0
  unsigned td1 = (unsigned)rows;               // tensor_dim1
  unsigned long long st0 = (unsigned long long)row_stride_elems;
  v8i_g g1;
  g1[0] = (int)d0;
  g1[1] = (int)((td0 & 0xffffu) << 16);                         // dim0[15:0]
  g1[2] = (int)((td0 >> 16) | ((td1 & 0xffffu) << 16));         // dim0[31:16]|dim1[15:0]
  g1[3] = (int)((td1 >> 16) | ((unsigned)rowlen_elems << 16));  // dim1[31:16]|tile_dim0
  g1[4] = (int)rows;                                            // tile_dim1 (tile_dim2=0)
  g1[5] = (int)(st0 & 0xffffffffu);                             // dim0_stride[31:0]
  g1[6] = (int)((st0 >> 32) & 0xffffu);                         // [47:32] (dim1_stride=0)
  g1[7] = 0;
  v4i_g z4 = {0, 0, 0, 0};
  v8i_g z8 = {0, 0, 0, 0, 0, 0, 0, 0};
  __builtin_amdgcn_tensor_load_to_lds(g0, g1, z4, z4, z8, 0);
}

__device__ __forceinline__ unsigned lds_off(const void* p) {
  return (unsigned)(size_t)(AS3 const void*)p;
}
#endif  // HAVE_TDM

// A-fragment (16x32 bf16). Lane l: row = l&15, K = (l>>4)*8 + {0..7} in v0..3
// and K = (l>>4)*8 + 16 + {0..7} in v4..7  (ISA 7.12.2, 16-bit A 16x32).
__device__ __forceinline__ v16bf load_fragA(const bf16_t* base, int stride,
                                            int row0, int k0, int lane) {
  int r = lane & 15, half = lane >> 4;
  const bf16_t* p = base + (size_t)(row0 + r) * stride + k0 + half * 8;
  union { v16bf v; uint4 q[2]; } u;
  u.q[0] = *(const uint4*)(p);
  u.q[1] = *(const uint4*)(p + 16);
  return u.v;
}

// B-fragment (32x16 bf16) from an N x K row-major matrix (i.e. B^T):
// Lane l: col n = l&15, K = (l>>4)*8 + {0..7} in v0..3 and (l>>4)*8 + 16 + {0..7} in v4..7.
__device__ __forceinline__ v16bf load_fragB(const bf16_t* base, int stride,
                                            int row0, int k0, int lane) {
  int r = lane & 15, half = lane >> 4;
  const bf16_t* p = base + (size_t)(row0 + r) * stride + k0 + half * 8;
  union { v16bf v; uint4 q[2]; } u;
  u.q[0] = *(const uint4*)(p);
  u.q[1] = *(const uint4*)(p + 16);
  return u.v;
}

__device__ __forceinline__ float gelu_exact(float x) {
  return 0.5f * x * (1.0f + erff(x * 0.70710678118654752f));
}

// Generic GEMM: Out[m,n] = epi( sum_k A[m,k] * W[n,k] ), A: MxK bf16 row-major,
// W: NxK bf16 row-major.  FLAGS: 1=bias 2=res(f32) 4=gelu 8=store-bf16.
// Block = 64(M) x 128(N), K-step 32; first tiles DMA'd by the TDM, K-loop tiles
// double-buffered with async global->LDS; 8 waves, 32x32 patch each.
template <int FLAGS>
__global__ __launch_bounds__(256) void gemm_wmma(
    const bf16_t* __restrict__ A, const bf16_t* __restrict__ W,
    const float* __restrict__ bias, const float* __restrict__ Res,
    float* __restrict__ outF, bf16_t* __restrict__ outB,
    int M, int N, int K) {
  // padded stride 40 elements (80B): conflict-free b128 LDS reads
  __shared__ __align__(16) bf16_t sA[2][64 * 40];
  __shared__ __align__(16) bf16_t sB[2][128 * 40];

  int tid = threadIdx.x;
  int lane = tid & 31;
  int wave = tid >> 5;
  int wm = wave >> 2;                   // 0..1
  int wn = wave & 3;                    // 0..3
  int mb = blockIdx.y * 64;
  int nb = blockIdx.x * 128;

  auto stage = [&](int buf, int k0) {
    {  // A tile: 64 rows x 32 K = 256 x 16B chunks, one per thread
      int row = tid >> 2, ch = (tid & 3) * 8;
      cp16(&sA[buf][row * 40 + ch], A + (size_t)(mb + row) * K + k0 + ch);
    }
#pragma unroll
    for (int j = 0; j < 2; j++) {  // B tile: 128 rows -> 512 chunks, 2/thread
      int idx = tid + j * 256;
      int row = idx >> 2, ch = (idx & 3) * 8;
      cp16(&sB[buf][row * 40 + ch], W + (size_t)(nb + row) * K + k0 + ch);
    }
  };

#if HAVE_TDM
  // Prologue tiles via Tensor Data Mover; one wave issues, all wait.
  if (wave == 0) {
    tdm_load_2d(A + (size_t)mb * K, lds_off(&sA[0][0]), 64, 32, K);
    tdm_load_2d(W + (size_t)nb * K, lds_off(&sB[0][0]), 128, 32, K);
    __builtin_amdgcn_s_wait_tensorcnt(0);
  }
  __syncthreads();
#else
  stage(0, 0);
  wait_stage();
#endif

  v8f acc[2][2] = {};
  int buf = 0;
  for (int k0 = 0; k0 < K; k0 += 32) {
    bool more = (k0 + 32 < K);
    if (more) stage(buf ^ 1, k0 + 32);

    const bf16_t* a_ = sA[buf];
    const bf16_t* b_ = sB[buf];
    v16bf a0 = load_fragA(a_, 40, wm * 32,      0, lane);
    v16bf a1 = load_fragA(a_, 40, wm * 32 + 16, 0, lane);
    v16bf b0 = load_fragB(b_, 40, wn * 32,      0, lane);
    v16bf b1 = load_fragB(b_, 40, wn * 32 + 16, 0, lane);
    acc[0][0] = WMMA_BF16(a0, b0, acc[0][0]);
    acc[0][1] = WMMA_BF16(a0, b1, acc[0][1]);
    acc[1][0] = WMMA_BF16(a1, b0, acc[1][0]);
    acc[1][1] = WMMA_BF16(a1, b1, acc[1][1]);

    if (more) { wait_stage(); buf ^= 1; }
  }

  int r = lane & 15, half = lane >> 4;
#pragma unroll
  for (int tn = 0; tn < 2; tn++) {
    int col = nb + wn * 32 + tn * 16 + r;
    float bv = 0.f;
    if constexpr (FLAGS & 1) bv = bias[col];
#pragma unroll
    for (int tm = 0; tm < 2; tm++) {
#pragma unroll
      for (int i = 0; i < 8; i++) {
        int row = mb + wm * 32 + tm * 16 + half * 8 + i;
        size_t idx = (size_t)row * N + col;
        float v = acc[tm][tn][i] + bv;
        if constexpr (FLAGS & 2) v += Res[idx];
        if constexpr (FLAGS & 4) v = gelu_exact(v);
        if constexpr (FLAGS & 8) outB[idx] = (bf16_t)v;
        else                     outF[idx] = v;
      }
    }
  }
}

// Fused linear attention per (b,h): scores = (q k^T)*scale (64x64) -> bf16 in
// LDS; o = scores @ v (64x32).  QKV layout: [token(2048), 1536] with q at col
// h*32+d, k at 512+h*32+d, v at 1024+h*32+d.  O: [token, h*32+d(512)].
__global__ __launch_bounds__(256) void attn_kernel(
    const bf16_t* __restrict__ QKV, bf16_t* __restrict__ O, float scale) {
  __shared__ __align__(16) bf16_t sS[64 * 64];   // scores
  __shared__ __align__(16) bf16_t sVT[32 * 64];  // v^T : [d][token]
  const int LDQ = 1536, HD = 512;
  int bh = blockIdx.x;
  int b = bh >> 4, h = bh & 15;
  size_t qbase = (size_t)(b * 64) * LDQ + h * 32;
  size_t kbase = qbase + 512;
  size_t vbase = qbase + 1024;
  size_t obase = (size_t)(b * 64) * HD + h * 32;
  int lane = threadIdx.x & 31;
  int wave = threadIdx.x >> 5;
  int r = lane & 15, half = lane >> 4;

  // Phase 1: 16 score tiles (4x4), each wave does two. K = 32 = one WMMA.
#pragma unroll
  for (int tt = 0; tt < 2; tt++) {
    int t = wave * 2 + tt;
    int tm = t >> 2, tn = t & 3;
    v16bf qf = load_fragA(QKV + qbase, LDQ, tm * 16, 0, lane);
    v16bf kf = load_fragB(QKV + kbase, LDQ, tn * 16, 0, lane);
    v8f c = {};
    c = WMMA_BF16(qf, kf, c);
#pragma unroll
    for (int i = 0; i < 8; i++) {
      int row = tm * 16 + half * 8 + i;
      int col = tn * 16 + r;
      sS[row * 64 + col] = (bf16_t)(c[i] * scale);
    }
  }
  // Stage v^T into LDS.
  for (int idx = threadIdx.x; idx < 64 * 32; idx += 256) {
    int t = idx >> 5, d = idx & 31;
    sVT[d * 64 + t] = QKV[vbase + (size_t)t * LDQ + d];
  }
  __syncthreads();

  // Phase 2: o tiles (4x2), one per wave; K = 64 = two WMMAs.
  int tm = wave >> 1, tn = wave & 1;
  v8f c = {};
#pragma unroll
  for (int k0 = 0; k0 < 64; k0 += 32) {
    v16bf af = load_fragA(sS, 64, tm * 16, k0, lane);
    v16bf bf = load_fragB(sVT, 64, tn * 16, k0, lane);
    c = WMMA_BF16(af, bf, c);
  }
#pragma unroll
  for (int i = 0; i < 8; i++) {
    int row = tm * 16 + half * 8 + i;  // query token
    int col = tn * 16 + r;             // d
    O[obase + (size_t)row * HD + col] = (bf16_t)c[i];
  }
}

__global__ __launch_bounds__(256) void ln_kernel(
    const float* __restrict__ Hb, const float* __restrict__ gamma,
    const float* __restrict__ beta, bf16_t* __restrict__ O) {
  int row = blockIdx.x;
  const float* x = Hb + (size_t)row * 1024;
  int lane = threadIdx.x & 31, wave = threadIdx.x >> 5;
  float s = 0.f, s2 = 0.f;
  for (int c = threadIdx.x; c < 1024; c += 256) {
    float v = x[c]; s += v; s2 += v * v;
  }
  for (int o = 16; o > 0; o >>= 1) {
    s += __shfl_down(s, o);
    s2 += __shfl_down(s2, o);
  }
  __shared__ float rs[8], rq[8], st[2];
  if (lane == 0) { rs[wave] = s; rq[wave] = s2; }
  __syncthreads();
  if (threadIdx.x == 0) {
    float a = 0.f, bb = 0.f;
    for (int i = 0; i < 8; i++) { a += rs[i]; bb += rq[i]; }
    float mu = a * (1.f / 1024.f);
    float var = bb * (1.f / 1024.f) - mu * mu;
    st[0] = mu; st[1] = rsqrtf(var + 1e-5f);
  }
  __syncthreads();
  float mu = st[0], inv = st[1];
  bf16_t* o_ = O + (size_t)row * 1024;
  for (int c = threadIdx.x; c < 1024; c += 256)
    o_[c] = (bf16_t)((x[c] - mu) * inv * gamma[c] + beta[c]);
}

__global__ void p_kernel(const float* __restrict__ ss, const float* __restrict__ w,
                         const float* __restrict__ b, float* __restrict__ p) {
  int i = blockIdx.x * blockDim.x + threadIdx.x;
  if (i < 1024) {
    float s = 0.f;
    for (int k = 0; k < 128; k++) s += ss[k] * w[i * 128 + k];
    p[i] = s + b[i];
  }
}

// xcat[(i*1024 + b*32 + n), kk], kk in [0,288): permuted xx views,
// column 256 = p, 257..287 zero-pad (K rounded to 288 for WMMA).
__global__ void xcat_kernel(const float* __restrict__ xx, const float* __restrict__ p,
                            bf16_t* __restrict__ xcat) {
  int idx = blockIdx.x * 256 + threadIdx.x;
  if (idx >= 3 * 1024 * 288) return;
  int token = idx / 288, kk = idx % 288;
  int i = token >> 10, bn = token & 1023;
  int b = bn >> 5, n = bn & 31;
  float v = 0.f;
  if (kk < 256) {
    int kd = kk >> 3, t = kk & 7;
    if (i == 0)      v = xx[(((t * 32 + b) * 32 + n) * 32) + kd];
    else if (i == 1) v = xx[(((t * 32 + n) * 32 + kd) * 32) + b];
    else             v = xx[(((t * 32 + kd) * 32 + b) * 32) + n];
  } else if (kk == 256) {
    v = p[bn];
  }
  xcat[idx] = (bf16_t)v;
}

__global__ void wpad_kernel(const float* __restrict__ w, bf16_t* __restrict__ o) {
  int idx = blockIdx.x * 256 + threadIdx.x;
  if (idx >= 1024 * 288) return;
  int c = idx / 288, kk = idx % 288;
  o[idx] = (bf16_t)(kk < 257 ? w[c * 257 + kk] : 0.f);
}

__global__ void cvt_kernel(const float* __restrict__ in, bf16_t* __restrict__ out, int n) {
  int i = blockIdx.x * 256 + threadIdx.x;
  if (i < n) out[i] = (bf16_t)in[i];
}

// a[b, j, c]: j<32 -> g[m1][b][j][c]; j>=32 -> g[m2][j-32][b][c] (transposed).
__global__ void build_a_kernel(const float* __restrict__ g, int m1, int m2,
                               float* __restrict__ aF, bf16_t* __restrict__ aB) {
  int idx = blockIdx.x * 256 + threadIdx.x;
  if (idx >= 2048 * 1024) return;
  int m = idx >> 10, c = idx & 1023;
  int b = m >> 6, j = m & 63;
  float v = (j < 32) ? g[((size_t)(m1 * 32 + b) * 32 + j) * 1024 + c]
                     : g[((size_t)(m2 * 32 + (j - 32)) * 32 + b) * 1024 + c];
  aF[idx] = v;
  aB[idx] = (bf16_t)v;
}

// g[m1][b][n][c] = cc[b,n,c]; g[m2][b][n][c] = cc[b,32+n,c]  (no un-transpose)
__global__ void scatter_kernel(const float* __restrict__ cc, float* __restrict__ g,
                               int m1, int m2) {
  int idx = blockIdx.x * 256 + threadIdx.x;
  if (idx >= 2048 * 1024) return;
  int m = idx >> 10, c = idx & 1023;
  int b = m >> 6, j = m & 63;
  if (j < 32) g[((size_t)(m1 * 32 + b) * 32 + j) * 1024 + c] = cc[idx];
  else        g[((size_t)(m2 * 32 + b) * 32 + (j - 32)) * 1024 + c] = cc[idx];
}

// ee[(b*96 + i*32 + n), c] = g[i][b][n][c]
__global__ void final_kernel(const float* __restrict__ g, float* __restrict__ out) {
  int idx = blockIdx.x * 256 + threadIdx.x;
  if (idx >= 3 * 1024 * 1024) return;
  int row = idx >> 10, c = idx & 1023;
  int b = row / 96, rr = row % 96;
  int i = rr >> 5, n = rr & 31;
  out[idx] = g[((size_t)(i * 32 + b) * 32 + n) * 1024 + c];
}

template <int FLAGS>
static inline void launch_gemm(const bf16_t* A, const bf16_t* W, const float* bias,
                               const float* Res, float* outF, bf16_t* outB,
                               int M, int N, int K, hipStream_t s) {
  dim3 grid(N / 128, M / 64);
  gemm_wmma<FLAGS><<<grid, 256, 0, s>>>(A, W, bias, Res, outF, outB, M, N, K);
}

extern "C" void kernel_launch(void* const* d_in, const int* in_sizes, int n_in,
                              void* d_out, int out_size, void* d_ws, size_t ws_size,
                              hipStream_t stream) {
  const float* xx     = (const float*)d_in[0];
  const float* ss     = (const float*)d_in[1];
  const float* li1_w  = (const float*)d_in[2];
  const float* li1_b  = (const float*)d_in[3];
  const float* li2_w  = (const float*)d_in[4];
  const float* li2_b  = (const float*)d_in[5];
  const float* Wq     = (const float*)d_in[6];
  const float* Wk     = (const float*)d_in[7];
  const float* Wv     = (const float*)d_in[8];
  const float* proj_w = (const float*)d_in[9];
  const float* proj_b = (const float*)d_in[10];
  const float* ln_g   = (const float*)d_in[11];
  const float* ln_b   = (const float*)d_in[12];
  const float* f1_w   = (const float*)d_in[13];
  const float* f1_b   = (const float*)d_in[14];
  const float* f2_w   = (const float*)d_in[15];
  const float* f2_b   = (const float*)d_in[16];
  float* out = (float*)d_out;

  char* ws = (char*)d_ws;
  size_t off = 0;
  auto alloc = [&](size_t bytes) -> void* {
    void* p = ws + off;
    off = (off + bytes + 255) & ~(size_t)255;
    return p;
  };
  float*  p_buf = (float*)alloc(1024 * 4);
  bf16_t* xcat  = (bf16_t*)alloc((size_t)3 * 1024 * 288 * 2);
  bf16_t* wpadb = (bf16_t*)alloc((size_t)1024 * 288 * 2);
  float*  gbuf  = (float*)alloc((size_t)3 * 1024 * 1024 * 4);
  bf16_t* wqkvb = (bf16_t*)alloc((size_t)1536 * 1024 * 2);
  bf16_t* wpb   = (bf16_t*)alloc((size_t)1024 * 512 * 2);
  bf16_t* wf1b  = (bf16_t*)alloc((size_t)4096 * 1024 * 2);
  bf16_t* wf2b  = (bf16_t*)alloc((size_t)1024 * 4096 * 2);
  float*  aF    = (float*)alloc((size_t)2048 * 1024 * 4);
  bf16_t* aB    = (bf16_t*)alloc((size_t)2048 * 1024 * 2);
  bf16_t* qkvB  = (bf16_t*)alloc((size_t)2048 * 1536 * 2);
  bf16_t* oB    = (bf16_t*)alloc((size_t)2048 * 512 * 2);
  float*  hF    = (float*)alloc((size_t)2048 * 1024 * 4);
  bf16_t* hnB   = (bf16_t*)alloc((size_t)2048 * 1024 * 2);
  bf16_t* midB  = (bf16_t*)alloc((size_t)2048 * 4096 * 2);
  float*  ccF   = (float*)alloc((size_t)2048 * 1024 * 4);
  (void)ws_size; (void)n_in; (void)in_sizes; (void)out_size;

  // ---- entry ----
  p_kernel<<<4, 256, 0, stream>>>(ss, li1_w, li1_b, p_buf);
  xcat_kernel<<<(3 * 1024 * 288 + 255) / 256, 256, 0, stream>>>(xx, p_buf, xcat);
  wpad_kernel<<<(1024 * 288 + 255) / 256, 256, 0, stream>>>(li2_w, wpadb);
  launch_gemm<1>(xcat, wpadb, li2_b, nullptr, gbuf, nullptr, 3072, 1024, 288, stream);

  const int pairs[3][2] = {{0, 1}, {1, 2}, {2, 0}};
  const float att_scale = 0.17677669529663687f;  // 32^-0.5

  for (int l = 0; l < 8; l++) {
    // per-layer weight staging fp32 -> bf16 (qkv stacked into one N=1536 matrix)
    cvt_kernel<<<2048, 256, 0, stream>>>(Wq + (size_t)l * 524288, wqkvb, 524288);
    cvt_kernel<<<2048, 256, 0, stream>>>(Wk + (size_t)l * 524288, wqkvb + 524288, 524288);
    cvt_kernel<<<2048, 256, 0, stream>>>(Wv + (size_t)l * 524288, wqkvb + 2 * 524288, 524288);
    cvt_kernel<<<2048, 256, 0, stream>>>(proj_w + (size_t)l * 524288, wpb, 524288);
    cvt_kernel<<<16384, 256, 0, stream>>>(f1_w + (size_t)l * 4194304, wf1b, 4194304);
    cvt_kernel<<<16384, 256, 0, stream>>>(f2_w + (size_t)l * 4194304, wf2b, 4194304);

    for (int pi = 0; pi < 3; pi++) {
      int m1 = pairs[pi][0], m2 = pairs[pi][1];
      build_a_kernel<<<(2048 * 1024) / 256, 256, 0, stream>>>(gbuf, m1, m2, aF, aB);

      // fused qkv projection: (2048x1024) @ (1024x1536), bf16 out
      launch_gemm<8>(aB, wqkvb, nullptr, nullptr, nullptr, qkvB,
                     2048, 1536, 1024, stream);

      attn_kernel<<<512, 256, 0, stream>>>(qkvB, oB, att_scale);

      // h = a + o @ proj_w^T + proj_b   (bias|res)
      launch_gemm<3>(oB, wpb, proj_b + (size_t)l * 1024, aF, hF, nullptr,
                     2048, 1024, 512, stream);

      ln_kernel<<<2048, 256, 0, stream>>>(hF, ln_g + (size_t)l * 1024,
                                          ln_b + (size_t)l * 1024, hnB);

      // mid = gelu(hn @ f1_w^T + f1_b)   (bias|gelu|bf16out)
      launch_gemm<13>(hnB, wf1b, f1_b + (size_t)l * 4096, nullptr, nullptr, midB,
                      2048, 4096, 1024, stream);
      // cc = h + mid @ f2_w^T + f2_b     (bias|res)
      launch_gemm<3>(midB, wf2b, f2_b + (size_t)l * 1024, hF, ccF, nullptr,
                     2048, 1024, 4096, stream);

      scatter_kernel<<<(2048 * 1024) / 256, 256, 0, stream>>>(ccF, gbuf, m1, m2);
    }
  }

  final_kernel<<<(3 * 1024 * 1024) / 256, 256, 0, stream>>>(gbuf, out);
}
